// RoPEAttention_58652073394883
// MI455X (gfx1250) — matrix operand order repaired and checked by reading, not compiled, not run on hardware
//
#include <hip/hip_runtime.h>

typedef __attribute__((ext_vector_type(16))) _Float16 v16h;
typedef __attribute__((ext_vector_type(8)))  float    v8f;
typedef __attribute__((ext_vector_type(4)))  unsigned int u32x4;
typedef __attribute__((ext_vector_type(8)))  int      i32x8;
typedef __attribute__((ext_vector_type(4)))  int      i32x4;

// ---------------------------------------------------------------------------
// WMMA helpers (CDNA5 v_wmma_f32_16x16x32_f16)
// ---------------------------------------------------------------------------
__device__ __forceinline__ v8f wmma_f16(v16h a, v16h b, v8f c) {
  return __builtin_amdgcn_wmma_f32_16x16x32_f16(
      /*neg_a=*/false, a, /*neg_b=*/false, b,
      /*c_mod=*/(short)0, c, /*reuse_a=*/false, /*reuse_b=*/false);
}

// A-matrix fragment (16x32 f16, row-major source, leading dim ld).
// ISA layout: lanes 0-15 row M=lane hold K {0..7, 16..23}; lanes 16-31 {8..15, 24..31}.
__device__ __forceinline__ v16h load_a16(const _Float16* __restrict__ p, int ld,
                                         int r0, int k0, int lane, int rmax) {
  int row = r0 + (lane & 15);
  if (row > rmax) row = rmax;
  int kb = k0 + ((lane >> 4) << 3);
  const _Float16* q = p + (size_t)row * ld + kb;
  v16h a;
#pragma unroll
  for (int j = 0; j < 8; ++j) { a[j] = q[j]; a[8 + j] = q[16 + j]; }
  return a;
}

// B-matrix fragment (32x16 f16). Element (kk, n) = p[n*ldn + kk].
// lane = column N (lanes 0-15 K {0..7, 16..23}, lanes 16-31 K {8..15, 24..31}).
__device__ __forceinline__ v16h load_b16(const _Float16* __restrict__ p, int ldn,
                                         int n0, int k0, int lane, int nmax, int kmax) {
  int n = n0 + (lane & 15);
  if (n > nmax) n = nmax;
  int kb = k0 + ((lane >> 4) << 3);
  const _Float16* q = p + (size_t)n * ldn;
  v16h b;
#pragma unroll
  for (int j = 0; j < 16; ++j) {
    int kk = kb + j + ((j >> 3) << 3);
    if (kk > kmax) kk = kmax;
    b[j] = q[kk];
  }
  return b;
}

// ---------------------------------------------------------------------------
// Tensor Data Mover: 2D f16 tile -> LDS (D# groups per ISA 08_async_tensor §8)
// group0: count=1 | lds_addr | global_addr[56:0] | type=2
// group1: data_size=2B | tensor_dim0/1 | tile_dim0/1 | dim0_stride
// (6-arg builtin form: g0, g1, g2, g3, g_extra, cpol)
// ---------------------------------------------------------------------------
__device__ __forceinline__ void tdm_load_2d(unsigned lds_off, const void* gptr,
                                            unsigned td0, unsigned td1,
                                            unsigned stride0,
                                            unsigned tile0, unsigned tile1) {
  unsigned long long ga = (unsigned long long)(uintptr_t)gptr;
  u32x4 g0;
  g0[0] = 1u;                                               // count=1, user mode
  g0[1] = lds_off;                                          // lds_addr (bytes)
  g0[2] = (unsigned)ga;                                     // global_addr[31:0]
  g0[3] = (unsigned)((ga >> 32) & 0x01ffffffu) | (2u << 30);// addr[56:32] | type=2
  i32x8 g1;
  g1[0] = (int)(1u << 16);                                  // data_size=1 (2 bytes)
  g1[1] = (int)((td0 & 0xffffu) << 16);                     // tensor_dim0[15:0]
  g1[2] = (int)((td0 >> 16) | ((td1 & 0xffffu) << 16));     // dim0 hi | dim1 lo
  g1[3] = (int)((td1 >> 16) | (tile0 << 16));               // dim1 hi | tile_dim0
  g1[4] = (int)(tile1 & 0xffffu);                           // tile_dim1 (tile_dim2=0)
  g1[5] = (int)stride0;                                     // tensor_dim0_stride lo
  g1[6] = 0;
  g1[7] = 0;
  i32x4 z4 = {0, 0, 0, 0};                                  // 2D: groups 2/3 unused
  i32x8 z8 = {0, 0, 0, 0, 0, 0, 0, 0};
  __builtin_amdgcn_tensor_load_to_lds(g0, g1, z4, z4, z8, 0);
}

// ---------------------------------------------------------------------------
// 1) f32 -> f16 conversion
// ---------------------------------------------------------------------------
__global__ void cvt_kernel(const float* __restrict__ in, _Float16* __restrict__ out, int n) {
  int i = blockIdx.x * blockDim.x + threadIdx.x;
  if (i < n) out[i] = (_Float16)in[i];
}

// ---------------------------------------------------------------------------
// 2) RoPE angle table: cos/sin [H=12][Np=576][F=32]
// ---------------------------------------------------------------------------
__global__ void rope_tab_kernel(const float* __restrict__ freqs,
                                float* __restrict__ cosT, float* __restrict__ sinT) {
  int gid = blockIdx.x * blockDim.x + threadIdx.x;
  if (gid >= 12 * 576 * 32) return;
  int f = gid & 31;
  int p = (gid >> 5) % 576;
  int h = gid / (576 * 32);
  float tx = (float)(p % 24);
  float ty = (float)(p / 24);
  int fi = h * 32 + f;
  float ang = tx * freqs[fi] + ty * freqs[384 + fi];
  float s, c;
  __sincosf(ang, &s, &c);
  cosT[gid] = c;
  sinT[gid] = s;
}

// ---------------------------------------------------------------------------
// 3) QKV GEMM [9232x768]@[768x2304] + RoPE epilogue.
//    One wave = one 16x16 output tile. q,k -> [B,H,N,HD]; v -> [B,H,HD,N].
// ---------------------------------------------------------------------------
__global__ __launch_bounds__(256) void qkv_rope_kernel(
    const _Float16* __restrict__ xf16, const _Float16* __restrict__ w16,
    const float* __restrict__ cosT, const float* __restrict__ sinT,
    _Float16* __restrict__ qf, _Float16* __restrict__ kf, _Float16* __restrict__ vT) {
  const int lane = threadIdx.x & 31;
  int w = blockIdx.x * 8 + (threadIdx.x >> 5);
  const int nTiles = 577 * 144;
  if (w >= nTiles) return;
  int rt = w / 144, ct = w % 144;
  int r0 = rt * 16, c0 = ct * 16;

  v8f acc = {};
#pragma unroll 2
  for (int kb = 0; kb < 768; kb += 32) {
    __builtin_prefetch(xf16 + (size_t)(r0 + (lane & 15)) * 768 + kb + 64, 0, 0);
    v16h a  = load_a16(xf16, 768, r0, kb, lane, 9231);
    v16h bb = load_b16(w16, 768, c0, kb, lane, 2303, 767);
    acc = wmma_f16(a, bb, acc);
  }

  int cl = lane & 15, hi = lane >> 4;
  int cc = c0 + cl;
  int which = cc / 768;          // 0=q 1=k 2=v (tile-uniform)
  int hcol = cc % 768;
  int h = hcol >> 6, d = hcol & 63;

#pragma unroll
  for (int r = 0; r < 8; ++r) {
    int row = r0 + r + hi * 8;   // flat b*577+n, always < 9232
    int b = row / 577, n = row % 577;
    float val = acc[r];
    float partner = __shfl_xor(val, 1, 32);   // pair lane, full EXEC
    if (which < 2 && n > 0) {                 // CLS (n==0) bypasses RoPE
      int f = d >> 1;
      int idx = (h * 576 + (n - 1)) * 32 + f;
      float c = cosT[idx], s = sinT[idx];
      val = ((d & 1) == 0) ? (val * c - partner * s) : (partner * s + val * c);
    }
    int bh = b * 12 + h;
    if (which == 0)      qf[((size_t)bh * 577 + n) * 64 + d] = (_Float16)val;
    else if (which == 1) kf[((size_t)bh * 577 + n) * 64 + d] = (_Float16)val;
    else                 vT[((size_t)bh * 64 + d) * 577 + n] = (_Float16)val;
  }
}

// ---------------------------------------------------------------------------
// 4) Flash attention. One block = one (b,h), 8 waves = 8 q-tiles of 16 rows.
//    Key blocks of 32 staged into LDS by wave 0 via TDM (double-buffered,
//    TENSORcnt + workgroup barrier). 4 WMMAs for S, online softmax, P->LDS
//    transpose, 4 WMMAs for O (HD=64). Output f16 [B,N,C].
// ---------------------------------------------------------------------------
__global__ __launch_bounds__(256) void flash_kernel(
    const _Float16* __restrict__ qf, const _Float16* __restrict__ kf,
    const _Float16* __restrict__ vT, _Float16* __restrict__ af) {
  __shared__ _Float16 skv[2][4096];  // per buffer: K tile 32x64 | V tile 64x32
  __shared__ _Float16 sp[8 * 512];   // per-wave private 16x32 P tile
  const int lane = threadIdx.x & 31;
  const int wave = threadIdx.x >> 5;
  const int blk = blockIdx.x;        // B*H*5
  const int bh = blk / 5;
  const int qg = blk % 5;
  int qt0 = qg * 8 + wave;
  bool dup = qt0 > 36;               // extra waves duplicate tile 36, skip store
  int qt = dup ? 36 : qt0;
  int n0 = qt * 16;
  const _Float16* qbh = qf + (size_t)bh * 577 * 64;
  const _Float16* kbh = kf + (size_t)bh * 577 * 64;
  const _Float16* vbh = vT + (size_t)bh * 64 * 577;
  int cl = lane & 15, hi = lane >> 4;
  _Float16* pp = sp + wave * 512;

  // q fragments (scaled by HD^-0.5 = 0.125); rows clamped at N-1.
  v16h aq0 = load_a16(qbh, 64, n0, 0, lane, 576);
  v16h aq1 = load_a16(qbh, 64, n0, 32, lane, 576);
#pragma unroll
  for (int j = 0; j < 16; ++j) { aq0[j] *= (_Float16)0.125f; aq1[j] *= (_Float16)0.125f; }

  float m[8], l[8];
#pragma unroll
  for (int r = 0; r < 8; ++r) { m[r] = -3.0e38f; l[r] = 0.0f; }
  v8f o0 = {}, o1 = {}, o2 = {}, o3 = {};

  // Prologue: stage key block 0 into buffer 0.
  if (wave == 0) {
    tdm_load_2d((unsigned)(uintptr_t)&skv[0][0],    kbh, 64u, 577u, 64u, 64u, 32u);
    tdm_load_2d((unsigned)(uintptr_t)&skv[0][2048], vbh, 577u, 64u, 577u, 32u, 64u);
    __builtin_amdgcn_s_wait_tensorcnt(0);
  }
  __syncthreads();

  for (int j = 0; j < 19; ++j) {
    int kb = j * 32;
    const _Float16* ldsK = skv[j & 1];         // [key_local 0..31][dim 0..63]
    const _Float16* ldsV = skv[j & 1] + 2048;  // [dim 0..63][key_local 0..31]

    // Kick off next block's DMA while computing this one.
    if (wave == 0 && j + 1 < 19) {
      int kn = kb + 32;
      tdm_load_2d((unsigned)(uintptr_t)&skv[(j + 1) & 1][0],
                  kbh + (size_t)kn * 64, 64u, (unsigned)(577 - kn), 64u, 64u, 32u);
      tdm_load_2d((unsigned)(uintptr_t)&skv[(j + 1) & 1][2048],
                  vbh + kn, (unsigned)(577 - kn), 64u, 577u, 32u, 64u);
    }

    // S = (q*scale) @ k^T for 32 keys, as two 16x16 tiles (K/LDS).
    v8f s0 = {}, s1 = {};
    v16h bk;
    bk = load_b16(ldsK, 64, 0,  0,  lane, 31, 63); s0 = wmma_f16(aq0, bk, s0);
    bk = load_b16(ldsK, 64, 0,  32, lane, 31, 63); s0 = wmma_f16(aq1, bk, s0);
    bk = load_b16(ldsK, 64, 16, 0,  lane, 31, 63); s1 = wmma_f16(aq0, bk, s1);
    bk = load_b16(ldsK, 64, 16, 32, lane, 31, 63); s1 = wmma_f16(aq1, bk, s1);

    int col0 = kb + cl, col1 = col0 + 16;
    if (col0 >= 577) {
#pragma unroll
      for (int r = 0; r < 8; ++r) s0[r] = -1.0e30f;
    }
    if (col1 >= 577) {
#pragma unroll
      for (int r = 0; r < 8; ++r) s1[r] = -1.0e30f;
    }

    float p0[8], p1[8];
#pragma unroll
    for (int r = 0; r < 8; ++r) {
      // row-wise reduce across the 16-lane half-groups (C layout)
      float mx = fmaxf(s0[r], s1[r]);
#pragma unroll
      for (int off = 1; off < 16; off <<= 1) mx = fmaxf(mx, __shfl_xor(mx, off, 32));
      float mn = fmaxf(m[r], mx);
      float corr = __expf(m[r] - mn);
      float e0 = __expf(s0[r] - mn);
      float e1 = __expf(s1[r] - mn);
      float rs = e0 + e1;
#pragma unroll
      for (int off = 1; off < 16; off <<= 1) rs += __shfl_xor(rs, off, 32);
      l[r] = l[r] * corr + rs;
      m[r] = mn;
      o0[r] *= corr; o1[r] *= corr; o2[r] *= corr; o3[r] *= corr;
      p0[r] = e0; p1[r] = e1;
    }

    // Transpose P (C layout -> A layout) through this wave's private LDS slice.
#pragma unroll
    for (int r = 0; r < 8; ++r) {
      int row = r + hi * 8;
      pp[row * 32 + cl]      = (_Float16)p0[r];
      pp[row * 32 + 16 + cl] = (_Float16)p1[r];
    }
    asm volatile("s_wait_dscnt 0" ::: "memory");  // same-wave DS RAW
    v16h ap = load_a16(pp, 32, 0, 0, lane, 15);

    // O += P @ V   (V tile in LDS, keys contiguous per dim row)
    v16h bv;
    bv = load_b16(ldsV, 32, 0,  0, lane, 63, 31); o0 = wmma_f16(ap, bv, o0);
    bv = load_b16(ldsV, 32, 16, 0, lane, 63, 31); o1 = wmma_f16(ap, bv, o1);
    bv = load_b16(ldsV, 32, 32, 0, lane, 63, 31); o2 = wmma_f16(ap, bv, o2);
    bv = load_b16(ldsV, 32, 48, 0, lane, 63, 31); o3 = wmma_f16(ap, bv, o3);

    // Next buffer ready (wave0 waits TDM) + all waves done reading this buffer.
    if (wave == 0) __builtin_amdgcn_s_wait_tensorcnt(0);
    __syncthreads();
  }

  if (!dup) {
    int b = bh / 12, h = bh % 12;
#pragma unroll
    for (int r = 0; r < 8; ++r) {
      int n = n0 + r + hi * 8;
      if (n >= 577) continue;
      float inv = 1.0f / l[r];
      size_t base = ((size_t)(b * 577 + n)) * 768 + h * 64 + cl;
      af[base + 0]  = (_Float16)(o0[r] * inv);
      af[base + 16] = (_Float16)(o1[r] * inv);
      af[base + 32] = (_Float16)(o2[r] * inv);
      af[base + 48] = (_Float16)(o3[r] * inv);
    }
  }
}

// ---------------------------------------------------------------------------
// 5) Output projection [9232x768]@[768x768] + bias -> f32
// ---------------------------------------------------------------------------
__global__ __launch_bounds__(256) void proj_kernel(
    const _Float16* __restrict__ af, const _Float16* __restrict__ wp,
    const float* __restrict__ bias, float* __restrict__ out) {
  const int lane = threadIdx.x & 31;
  int w = blockIdx.x * 8 + (threadIdx.x >> 5);
  if (w >= 577 * 48) return;
  int rt = w / 48, ct = w % 48;
  int r0 = rt * 16, c0 = ct * 16;

  v8f acc = {};
#pragma unroll 2
  for (int kb = 0; kb < 768; kb += 32) {
    __builtin_prefetch(af + (size_t)(r0 + (lane & 15)) * 768 + kb + 64, 0, 0);
    v16h a  = load_a16(af, 768, r0, kb, lane, 9231);
    v16h bb = load_b16(wp, 768, c0, kb, lane, 767, 767);
    acc = wmma_f16(a, bb, acc);
  }
  int cl = lane & 15, hi = lane >> 4;
  float bv = bias[c0 + cl];
#pragma unroll
  for (int r = 0; r < 8; ++r) {
    int row = r0 + r + hi * 8;   // 9232 = 577*16 -> always in range
    out[(size_t)row * 768 + c0 + cl] = acc[r] + bv;
  }
}

// ---------------------------------------------------------------------------
// Launcher
// ---------------------------------------------------------------------------
extern "C" void kernel_launch(void* const* d_in, const int* in_sizes, int n_in,
                              void* d_out, int out_size, void* d_ws, size_t ws_size,
                              hipStream_t stream) {
  (void)in_sizes; (void)n_in; (void)out_size; (void)ws_size;
  const float* x      = (const float*)d_in[0];
  const float* qkv_w  = (const float*)d_in[1];
  const float* proj_w = (const float*)d_in[2];
  const float* proj_b = (const float*)d_in[3];
  const float* freqs  = (const float*)d_in[4];
  float* out = (float*)d_out;

  char* ws = (char*)d_ws;
  size_t off = 0;
  auto wsalloc = [&](size_t bytes) -> void* {
    void* p = ws + off;
    off += (bytes + 255) & ~(size_t)255;
    return p;
  };
  _Float16* xf16 = (_Float16*)wsalloc((size_t)9232 * 768 * 2);
  _Float16* wq16 = (_Float16*)wsalloc((size_t)2304 * 768 * 2);
  _Float16* wp16 = (_Float16*)wsalloc((size_t)768 * 768 * 2);
  float*    cosT = (float*)wsalloc((size_t)12 * 576 * 32 * 4);
  float*    sinT = (float*)wsalloc((size_t)12 * 576 * 32 * 4);
  _Float16* qf   = (_Float16*)wsalloc((size_t)16 * 12 * 577 * 64 * 2);
  _Float16* kf   = (_Float16*)wsalloc((size_t)16 * 12 * 577 * 64 * 2);
  _Float16* vT   = (_Float16*)wsalloc((size_t)16 * 12 * 577 * 64 * 2);
  _Float16* af   = (_Float16*)wsalloc((size_t)9232 * 768 * 2);

  cvt_kernel<<<(9232 * 768 + 255) / 256, 256, 0, stream>>>(x, xf16, 9232 * 768);
  cvt_kernel<<<(2304 * 768 + 255) / 256, 256, 0, stream>>>(qkv_w, wq16, 2304 * 768);
  cvt_kernel<<<(768 * 768 + 255) / 256, 256, 0, stream>>>(proj_w, wp16, 768 * 768);
  rope_tab_kernel<<<(12 * 576 * 32 + 255) / 256, 256, 0, stream>>>(freqs, cosT, sinT);
  qkv_rope_kernel<<<(577 * 144 + 7) / 8, 256, 0, stream>>>(xf16, wq16, cosT, sinT, qf, kf, vT);
  flash_kernel<<<16 * 12 * 5, 256, 0, stream>>>(qf, kf, vT, af);
  proj_kernel<<<(577 * 48 + 7) / 8, 256, 0, stream>>>(af, wp16, proj_b, out);
}
